// RNN_43250320670779
// MI455X (gfx1250) — matrix operand order repaired and checked by reading, not compiled, not run on hardware
//
#include <hip/hip_runtime.h>
#include <hip/hip_bf16.h>

// ---------------------------------------------------------------------------
// 2-layer masked SimpleRNN for MI455X (gfx1250, wave32, WMMA).
//   xproj_gemm : gather + [B*T,128]x[128,128] GEMM via v_wmma_f32_16x16x32_bf16
//   rnn_scan   : latency-optimized sequential scan. One barrier/step,
//                f32 state in VGPRs, bf16 state ping-pong in LDS,
//                Wh fragments resident in VGPRs, ballot-based masking.
// ---------------------------------------------------------------------------

typedef __bf16        v16bf __attribute__((ext_vector_type(16)));
typedef __bf16        v2bf  __attribute__((ext_vector_type(2)));
typedef float         v8f   __attribute__((ext_vector_type(8)));
typedef float         v2f   __attribute__((ext_vector_type(2)));
typedef unsigned int  v8u   __attribute__((ext_vector_type(8)));

#define RNN_B 32
#define RNN_H 128

// ---- f32 -> bf16 via native conversion (backend picks v_cvt_pk_bf16_f32) --
static __device__ __forceinline__ unsigned int pack2(float a, float b) {
  v2f f = {a, b};
  v2bf r = __builtin_convertvector(f, v2bf);
  return __builtin_bit_cast(unsigned int, r);
}
static __device__ __forceinline__ unsigned short f2bf(float f) {
  __bf16 h = (__bf16)f;
  return __builtin_bit_cast(unsigned short, h);
}
static __device__ __forceinline__ float fast_tanh(float x) {
#if __has_builtin(__builtin_amdgcn_tanhf)
  return __builtin_amdgcn_tanhf(x);           // v_tanh_f32 (1 trans op)
#elif __has_builtin(__builtin_amdgcn_tanh_f32)
  return __builtin_amdgcn_tanh_f32(x);
#else
  return tanhf(x);
#endif
}
static __device__ __forceinline__ unsigned int ballot32(int pred) {
#if __has_builtin(__builtin_amdgcn_ballot_w32)
  return __builtin_amdgcn_ballot_w32(pred != 0);
#else
  return (unsigned int)__ballot(pred);
#endif
}
static __device__ __forceinline__ v16bf mkfrag(uint4 lo, uint4 hi) {
  v8u u;
  u[0]=lo.x; u[1]=lo.y; u[2]=lo.z; u[3]=lo.w;
  u[4]=hi.x; u[5]=hi.y; u[6]=hi.z; u[7]=hi.w;
  return __builtin_bit_cast(v16bf, u);
}
static __device__ __forceinline__ v8f wmma_bf16(v16bf a, v16bf b, v8f c) {
  return __builtin_amdgcn_wmma_f32_16x16x32_bf16(false, a, false, b, (short)0, c,
                                                 false, false);
}

// ---------------------------------------------------------------------------
// xproj = (gather ? emb[tokens] : inRows) @ W + bias
// in rows: r = t*32+b (row-major [r][128]); out layout: [T, H, B] (t, col, b)
// grid: (B*T)/128 blocks, 256 threads (8 waves); wave -> one 16-row M tile.
// ---------------------------------------------------------------------------
__global__ __launch_bounds__(256) void xproj_gemm(
    const float* __restrict__ inRows,
    const int*   __restrict__ tokens,
    const float* __restrict__ emb,
    const float* __restrict__ W,      // [128,128] row-major (k, n)
    const float* __restrict__ bias,   // [128]
    float*       __restrict__ out,    // [T,128,32]
    int gather, int T)
{
  __shared__ __align__(16) unsigned int Wb[RNN_H * (RNN_H / 2)]; // bf16 packed
  const int tid = threadIdx.x;

  for (int i = tid; i < RNN_H * (RNN_H / 2); i += 256) {
    int c = i >> 6, k = (i & 63) << 1;
    Wb[i] = pack2(W[k * RNN_H + c], W[(k + 1) * RNN_H + c]);
  }
  __syncthreads();

  const int  wave  = tid >> 5, lane = tid & 31;
  const int  laneM = lane & 15;
  const int  khalf = (lane < 16) ? 0 : 8;
  const long rowBase = (long)blockIdx.x * 128 + wave * 16;  // 16-row tile, one t
  const long myRow   = rowBase + laneM;

  const float* src;
  if (gather) {
    int b   = (int)(myRow & 31);
    int t   = (int)(myRow >> 5);
    int tok = tokens[b * T + t];
    src = emb + (long)tok * RNN_H;
  } else {
    src = inRows + myRow * RNN_H;
  }

  // A fragments: lane holds row M=laneM, K runs [kb..kb+7] and [kb+16..kb+23]
  v16bf A[4];
#pragma unroll
  for (int kt = 0; kt < 4; ++kt) {
    int kb = kt * 32 + khalf;
    float4 a0 = *(const float4*)(src + kb);
    float4 a1 = *(const float4*)(src + kb + 4);
    float4 a2 = *(const float4*)(src + kb + 16);
    float4 a3 = *(const float4*)(src + kb + 20);
    v8u u;
    u[0] = pack2(a0.x, a0.y); u[1] = pack2(a0.z, a0.w);
    u[2] = pack2(a1.x, a1.y); u[3] = pack2(a1.z, a1.w);
    u[4] = pack2(a2.x, a2.y); u[5] = pack2(a2.z, a2.w);
    u[6] = pack2(a3.x, a3.y); u[7] = pack2(a3.z, a3.w);
    A[kt] = __builtin_bit_cast(v16bf, u);
  }

  const int  tT = (int)(rowBase >> 5);            // whole tile shares this t
  const int  b0 = (int)(rowBase & 31) + khalf;    // batch of acc[0]

#pragma unroll
  for (int nt = 0; nt < 8; ++nt) {
    v8f acc = {0.f, 0.f, 0.f, 0.f, 0.f, 0.f, 0.f, 0.f};
#pragma unroll
    for (int kt = 0; kt < 4; ++kt) {
      int K = kt * 32 + khalf;                    // B: N col per lane
      const uint4* p = (const uint4*)&Wb[(nt * 16 + laneM) * 64 + (K >> 1)];
      v16bf Bf = mkfrag(p[0], p[2]);              // K [K..K+7] and [K+16..K+23]
      acc = wmma_bf16(A[kt], Bf, acc);
    }
    int   col = nt * 16 + laneM;
    float bv  = bias[col];
    float4 lo{acc[0] + bv, acc[1] + bv, acc[2] + bv, acc[3] + bv};
    float4 hi{acc[4] + bv, acc[5] + bv, acc[6] + bv, acc[7] + bv};
    float* op = out + ((long)tT * RNN_H + col) * RNN_B + b0;  // 16B aligned
    *(float4*)op       = lo;
    *(float4*)(op + 4) = hi;
  }
}

// ---------------------------------------------------------------------------
// Sequential masked scan: h = mask ? tanh(xp[t] + h@Wh) : h  (all 32 batches)
// One block, 8 waves; wave owns a 16-column N slice; Wh frags in VGPRs;
// f32 state in VGPRs (lane-private carry); bf16 state ping-pong in LDS;
// per-step mask from v_ballot (wave32), no LDS mask traffic.
// ---------------------------------------------------------------------------
__global__ __launch_bounds__(256) void rnn_scan(
    const float* __restrict__ xp,      // [T, 128, 32]  (t, h, b)
    const float* __restrict__ Wh,      // [128,128]
    const int*   __restrict__ tokens,  // [32, T]
    float*       __restrict__ out,
    int out_bt, int T)                 // out_bt: 1 -> [B,T,H], 0 -> [T,B,H]
{
  __shared__ __align__(16) unsigned int   WhB[RNN_H * (RNN_H / 2)];
  __shared__ __align__(16) unsigned short Sb[2][RNN_B * RNN_H];  // bf16 state

  const int tid = threadIdx.x;
  for (int i = tid; i < RNN_H * (RNN_H / 2); i += 256) {
    int c = i >> 6, k = (i & 63) << 1;
    WhB[i] = pack2(Wh[k * RNN_H + c], Wh[(k + 1) * RNN_H + c]);
  }
  for (int i = tid; i < RNN_B * RNN_H; i += 256) Sb[0][i] = 0;
  __syncthreads();

  const int wave  = tid >> 5, lane = tid & 31;
  const int laneM = lane & 15;
  const int khalf = (lane < 16) ? 0 : 8;
  const int col   = wave * 16 + laneM;

  // Loop-invariant Wh B-fragments, resident in VGPRs for all timesteps.
  v16bf Bf[4];
#pragma unroll
  for (int kt = 0; kt < 4; ++kt) {
    int K = kt * 32 + khalf;
    const uint4* p = (const uint4*)&WhB[(wave * 16 + laneM) * 64 + (K >> 1)];
    Bf[kt] = mkfrag(p[0], p[2]);
  }

  // Loop-invariant 32-bit address pieces (SGPR-base + VGPR-offset stores).
  const int xcol = col * RNN_B;                 // xp column base
  const int tStepOut = out_bt ? RNN_H : RNN_B * RNN_H;   // 128 : 4096
  int outBase[2][8];
  int sbIdx[2][8];
  float hreg[2][8];                             // lane-private f32 state
#pragma unroll
  for (int m = 0; m < 2; ++m)
#pragma unroll
    for (int r = 0; r < 8; ++r) {
      int row = m * 16 + r + khalf;
      outBase[m][r] = out_bt ? (row * T * RNN_H + col) : (row * RNN_H + col);
      sbIdx[m][r]   = row * RNN_H + col;
      hreg[m][r]    = 0.f;
    }

  for (int t = 0; t < T; ++t) {
    const int cur = t & 1, nxt = cur ^ 1;

    // Per-step mask: wave32 ballot over this step's 32 tokens (rows 0..31).
    int tok = tokens[lane * T + t];
    unsigned int mrows = ballot32(tok != 0);

    // xp for this step: [t][col][b], lane's 16 batches = 4x b128 loads
    const float* xb = xp + (size_t)t * (RNN_B * RNN_H) + xcol;
    float4 x0 = *(const float4*)(xb + khalf);
    float4 x1 = *(const float4*)(xb + khalf + 4);
    float4 x2 = *(const float4*)(xb + 16 + khalf);
    float4 x3 = *(const float4*)(xb + 16 + khalf + 4);
    float xpv[2][8] = {{x0.x, x0.y, x0.z, x0.w, x1.x, x1.y, x1.z, x1.w},
                       {x2.x, x2.y, x2.z, x2.w, x3.x, x3.y, x3.z, x3.w}};
    if (t + 1 < T)
      __builtin_prefetch(xp + (size_t)(t + 1) * (RNN_B * RNN_H) + tid * 16, 0, 1);

    // A fragments from bf16 state h_{t-1}
    v16bf Af[2][4];
#pragma unroll
    for (int m = 0; m < 2; ++m)
#pragma unroll
      for (int kt = 0; kt < 4; ++kt) {
        int base = (m * 16 + laneM) * RNN_H + kt * 32 + khalf;
        uint4 lo = *(const uint4*)&Sb[cur][base];
        uint4 hi = *(const uint4*)&Sb[cur][base + 16];
        Af[m][kt] = mkfrag(lo, hi);
      }

    // 4 independent WMMA chains (depth 2) -> more ILP on the latency path
    v8f a0e = {0,0,0,0,0,0,0,0}, a0o = {0,0,0,0,0,0,0,0};
    v8f a1e = {0,0,0,0,0,0,0,0}, a1o = {0,0,0,0,0,0,0,0};
    a0e = wmma_bf16(Af[0][0], Bf[0], a0e);
    a0o = wmma_bf16(Af[0][1], Bf[1], a0o);
    a1e = wmma_bf16(Af[1][0], Bf[0], a1e);
    a1o = wmma_bf16(Af[1][1], Bf[1], a1o);
    a0e = wmma_bf16(Af[0][2], Bf[2], a0e);
    a0o = wmma_bf16(Af[0][3], Bf[3], a0o);
    a1e = wmma_bf16(Af[1][2], Bf[2], a1e);
    a1o = wmma_bf16(Af[1][3], Bf[3], a1o);
    v8f acc0 = a0e + a0o;
    v8f acc1 = a1e + a1o;

    float* op = out + (size_t)t * tStepOut;     // uniform -> SGPR base
#pragma unroll
    for (int m = 0; m < 2; ++m) {
#pragma unroll
      for (int r = 0; r < 8; ++r) {
        int   row = m * 16 + r + khalf;
        float v   = (m ? acc1[r] : acc0[r]) + xpv[m][r];
        v = fast_tanh(v);
        v = ((mrows >> row) & 1u) ? v : hreg[m][r];  // masked carry (f32 VGPR)
        hreg[m][r] = v;
        Sb[nxt][sbIdx[m][r]] = f2bf(v);              // publish for next step
        op[outBase[m][r]] = v;                       // 32-bit voffset store
      }
    }
    __syncthreads();   // single barrier/step: publish Sb[nxt]
  }
}

// ---------------------------------------------------------------------------
extern "C" void kernel_launch(void* const* d_in, const int* in_sizes, int n_in,
                              void* d_out, int out_size, void* d_ws, size_t ws_size,
                              hipStream_t stream) {
  (void)in_sizes; (void)n_in; (void)out_size; (void)ws_size;
  const int*   tokens = (const int*)  d_in[0];
  const float* emb    = (const float*)d_in[1];
  const float* Wx0    = (const float*)d_in[2];
  const float* Wh0    = (const float*)d_in[3];
  const float* b0     = (const float*)d_in[4];
  const float* Wx1    = (const float*)d_in[5];
  const float* Wh1    = (const float*)d_in[6];
  const float* b1     = (const float*)d_in[7];
  float* out = (float*)d_out;

  const int    T = 2048;
  const size_t n = (size_t)RNN_B * T * RNN_H;
  float* xpbuf = (float*)d_ws;        // [T,H,B] projections (reused per layer)
  float* y0    = xpbuf + n;           // [T,B,H] layer-0 outputs (GEMM-1 input)

  const int blocks = (RNN_B * T) / 128;   // 512

  // layer 0: gather + input projection, then masked scan
  xproj_gemm<<<blocks, 256, 0, stream>>>(nullptr, tokens, emb, Wx0, b0, xpbuf, 1, T);
  rnn_scan  <<<1,      256, 0, stream>>>(xpbuf, Wh0, tokens, y0, 0, T);

  // layer 1: input projection of y0, then masked scan into [B,T,H] output
  xproj_gemm<<<blocks, 256, 0, stream>>>(y0, nullptr, nullptr, Wx1, b1, xpbuf, 0, T);
  rnn_scan  <<<1,      256, 0, stream>>>(xpbuf, Wh1, tokens, out, 1, T);
}
